// DTA_Model_59184649339420
// MI455X (gfx1250) — matrix operand order repaired and checked by reading, not compiled, not run on hardware
//
#include <hip/hip_runtime.h>
#include <hip/hip_bf16.h>

typedef __attribute__((ext_vector_type(16))) __bf16 v16bf;
typedef __attribute__((ext_vector_type(8)))  __bf16 v8bf;
typedef __attribute__((ext_vector_type(4)))  __bf16 v4bf;
typedef __attribute__((ext_vector_type(8)))  float  v8f;

#define NN   131072
#define EE   524288
#define BB   4096
#define HH   256
#define NFD  29
#define EFD  7
#define BN_INV 0.99999500003749969f

// ---------------- elementwise helpers ----------------
__global__ void k_fill(float* p, float v, int n) {
    int i = blockIdx.x * 256 + threadIdx.x;
    if (i < n) p[i] = v;
}

__global__ void k_copy4(const float4* __restrict__ s, float4* __restrict__ d, long n4) {
    long i = (long)blockIdx.x * 256 + threadIdx.x;
    if (i < n4) d[i] = s[i];
}

// x [N,29] -> d [N,32] zero-padded
__global__ void k_pad29(const float* __restrict__ x, float* __restrict__ d) {
    long idx = (long)blockIdx.x * 256 + threadIdx.x;
    if (idx >= (long)NN * 32) return;
    int r = (int)(idx >> 5), c = (int)(idx & 31);
    d[idx] = (c < NFD) ? x[(long)r * NFD + c] : 0.0f;
}

// W [K,Nout] f32 -> Wt [Nout,Kpad] bf16 (zero-padded K)
__global__ void k_wt(const float* __restrict__ W, __bf16* __restrict__ Wt,
                     int K, int Kpad, int Nout) {
    int idx = blockIdx.x * 256 + threadIdx.x;
    if (idx >= Nout * Kpad) return;
    int n = idx / Kpad, k = idx % Kpad;
    float v = (k < K) ? W[k * Nout + n] : 0.0f;
    Wt[idx] = (__bf16)v;
}

// ---------------- GINE edge message + scatter ----------------
// layer 0: feature dim 29 (agg padded to lda=32). One wave per edge.
__global__ void k_edge0(const float* __restrict__ x, const int* __restrict__ ei,
                        const float* __restrict__ ea, const float* __restrict__ We,
                        const float* __restrict__ beo, float* agg) {
    int wave = (blockIdx.x * blockDim.x + threadIdx.x) >> 5;
    int lane = threadIdx.x & 31;
    if (wave >= EE) return;
    int src = ei[wave], dst = ei[EE + wave];
    if (lane < NFD) {
        float lin = beo[lane];
        #pragma unroll
        for (int j = 0; j < EFD; ++j)
            lin += ea[(long)wave * EFD + j] * We[j * NFD + lane];
        float m = x[(long)src * NFD + lane] + lin;
        m = fmaxf(m, 0.0f);
        atomicAdd(&agg[(long)dst * 32 + lane], m);
    }
}

// recurrent layers: feature dim 256. One wave per edge, 8 features/lane.
__global__ void k_edge(const float* __restrict__ h, const int* __restrict__ ei,
                       const float* __restrict__ ea, const float* __restrict__ We,
                       const float* __restrict__ beo, float* agg) {
    int wave = (blockIdx.x * blockDim.x + threadIdx.x) >> 5;
    int lane = threadIdx.x & 31;
    if (wave >= EE) return;
    int src = ei[wave], dst = ei[EE + wave];
    float ev[EFD];
    #pragma unroll
    for (int j = 0; j < EFD; ++j) ev[j] = ea[(long)wave * EFD + j];
    #pragma unroll
    for (int t = 0; t < 8; ++t) {
        int f = lane + 32 * t;
        float lin = beo[f];
        #pragma unroll
        for (int j = 0; j < EFD; ++j) lin += ev[j] * We[j * HH + f];
        float m = h[(long)src * HH + f] + lin;
        m = fmaxf(m, 0.0f);
        atomicAdd(&agg[(long)dst * HH + f], m);
    }
}

// ---------------- WMMA GEMM ----------------
// C = epilogue(A[M,Kpad] @ Wt^T + bias). Requirements (guaranteed by launcher):
//   M % 32 == 0, Nout % 64 == 0, Kpad % KC_ == 0, lda % 4 == 0, A 16B-aligned.
// 8 waves arranged 2x4 (16x16 WMMA tiles) -> 32x64 output tile per workgroup.
template<int KC_>
__global__ __launch_bounds__(256) void k_gemm(
    const float* __restrict__ A, int lda, int Kpad,
    const __bf16* __restrict__ Wt,
    const float* __restrict__ bias, const float* __restrict__ gamma,
    const float* __restrict__ beta, int do_relu, float* __restrict__ C, int ldc)
{
    constexpr int KCP = KC_ + 8;     // padded LDS stride (bf16), keeps 16B alignment
    __shared__ __bf16 Asub[32 * KCP];
    __shared__ __bf16 Bsub[64 * KCP];

    int tid  = threadIdx.x;
    int lane = tid & 31;
    int w    = tid >> 5;
    int wr   = w >> 2;               // 0..1 wave row-tile
    int wc   = w & 3;                // 0..3 wave col-tile
    int rbase = blockIdx.x * 32;
    int cbase = blockIdx.y * 64;
    int m16 = lane & 15;
    int hi  = lane >> 4;

    v8f acc = {};

    for (int k0 = 0; k0 < Kpad; k0 += KC_) {
        // stage A tile: float4 global loads -> packed bf16 (b64 ds stores)
        constexpr int KV = KC_ / 4;
        #pragma unroll
        for (int t0 = 0; t0 < 32 * KV; t0 += 256) {
            int t = t0 + tid;
            int r = t / KV, kv = t % KV;
            float4 s = *(const float4*)(A + (long)(rbase + r) * lda + k0 + kv * 4);
            v4bf d;
            d[0] = (__bf16)s.x; d[1] = (__bf16)s.y;
            d[2] = (__bf16)s.z; d[3] = (__bf16)s.w;
            *(v4bf*)&Asub[r * KCP + kv * 4] = d;
        }
        // stage B tile: 8xbf16 vector loads/stores (b128 both sides)
        constexpr int KW = KC_ / 8;
        #pragma unroll
        for (int t0 = 0; t0 < 64 * KW; t0 += 256) {
            int t = t0 + tid;
            int n = t / KW, kw = t % KW;
            v8bf s = *(const v8bf*)(Wt + (long)(cbase + n) * Kpad + k0 + kw * 8);
            *(v8bf*)&Bsub[n * KCP + kw * 8] = s;
        }
        __syncthreads();

        #pragma unroll
        for (int kk = 0; kk < KC_; kk += 32) {
            // A fragment 16x32: lane<16 -> K=0..15; lane>=16 -> K=16..31
            v16bf af;
            const __bf16* ap = &Asub[(wr * 16 + m16) * KCP + kk + hi * 16];
            #pragma unroll
            for (int i = 0; i < 8; ++i) { af[i] = ap[i]; af[i + 8] = ap[8 + i]; }
            // B fragment 32x16: lane<16 -> K=0..15; lane>=16 -> K=16..31; N=lane&15
            v16bf bfm;
            const __bf16* bp = &Bsub[(wc * 16 + m16) * KCP + kk + hi * 16];
            #pragma unroll
            for (int i = 0; i < 16; ++i) bfm[i] = bp[i];
            acc = __builtin_amdgcn_wmma_f32_16x16x32_bf16(
                false, af, false, bfm, (short)0, acc, false, false);
        }
        __syncthreads();
    }

    // epilogue: bias (+ optional BN-eval scale/shift) (+ optional ReLU)
    int col = cbase + wc * 16 + m16;
    #pragma unroll
    for (int r = 0; r < 8; ++r) {
        int row = rbase + wr * 16 + r + 8 * hi;
        float v = acc[r] + bias[col];
        if (gamma) v = gamma[col] * (v * BN_INV) + beta[col];
        if (do_relu) v = fmaxf(v, 0.0f);
        C[(long)row * ldc + col] = v;
    }
}

// ---------------- pooling & head ----------------
__global__ void k_pool_sum(const float* __restrict__ h, const int* __restrict__ batch,
                           float* sums, float* counts) {
    int wave = (blockIdx.x * blockDim.x + threadIdx.x) >> 5;
    int lane = threadIdx.x & 31;
    if (wave >= NN) return;
    int b = batch[wave];
    #pragma unroll
    for (int t = 0; t < 8; ++t) {
        int f = lane + 32 * t;
        atomicAdd(&sums[(long)b * HH + f], h[(long)wave * HH + f]);
    }
    if (lane == 0) atomicAdd(&counts[b], 1.0f);
}

__global__ void k_pool_div(const float* __restrict__ sums, const float* __restrict__ counts,
                           float* __restrict__ z) {
    int idx = blockIdx.x * 256 + threadIdx.x;
    if (idx >= BB * HH) return;
    int b = idx / HH, f = idx % HH;
    float c = fmaxf(counts[b], 1.0f);
    z[(long)b * 768 + f] = sums[idx] / c;   // drug_graph -> z[:, 0:256]
}

__global__ void k_final(const float* __restrict__ z2, const float* __restrict__ hW3,
                        const float* __restrict__ hb3, float* __restrict__ out) {
    int wave = (blockIdx.x * blockDim.x + threadIdx.x) >> 5;
    int lane = threadIdx.x & 31;
    if (wave >= BB) return;
    float v = 0.0f;
    #pragma unroll
    for (int t = 0; t < 4; ++t) {
        int j = lane + 32 * t;
        v += z2[(long)wave * 128 + j] * hW3[j];
    }
    #pragma unroll
    for (int off = 16; off > 0; off >>= 1) v += __shfl_xor(v, off, 32);
    if (lane == 0) out[wave] = v + hb3[0];
}

// ---------------- launch ----------------
#define GEMM(KCV, A, lda, M, Kpad, Wt, Nout, bias, gamma, beta, relu, C, ldc)       \
    k_gemm<KCV><<<dim3((M) / 32, (Nout) / 64), 256, 0, stream>>>(                   \
        A, lda, Kpad, Wt, bias, gamma, beta, relu, C, ldc)

extern "C" void kernel_launch(void* const* d_in, const int* in_sizes, int n_in,
                              void* d_out, int out_size, void* d_ws, size_t ws_size,
                              hipStream_t stream) {
    const float* x     = (const float*)d_in[0];
    const int*   ei    = (const int*)d_in[1];
    const float* ea    = (const float*)d_in[2];
    const int*   batch = (const int*)d_in[3];
    const float* fp    = (const float*)d_in[4];
    const float* te    = (const float*)d_in[5];
    const float* We0   = (const float*)d_in[6];
    const float* beo0  = (const float*)d_in[7];
    const float* W1_0  = (const float*)d_in[8];
    const float* b1_0  = (const float*)d_in[9];
    const float* g1_0  = (const float*)d_in[10];
    const float* bt1_0 = (const float*)d_in[11];
    const float* W2_0  = (const float*)d_in[12];
    const float* b2_0  = (const float*)d_in[13];
    const float* gbn0  = (const float*)d_in[14];
    const float* bbn0  = (const float*)d_in[15];
    const float* We_r  = (const float*)d_in[16];
    const float* beo_r = (const float*)d_in[17];
    const float* W1_r  = (const float*)d_in[18];
    const float* b1_r  = (const float*)d_in[19];
    const float* g1_r  = (const float*)d_in[20];
    const float* bt1_r = (const float*)d_in[21];
    const float* W2_r  = (const float*)d_in[22];
    const float* b2_r  = (const float*)d_in[23];
    const float* gbn_r = (const float*)d_in[24];
    const float* bbn_r = (const float*)d_in[25];
    const float* fpW   = (const float*)d_in[26];
    const float* fpb   = (const float*)d_in[27];
    const float* pW1   = (const float*)d_in[28];
    const float* pb1   = (const float*)d_in[29];
    const float* pW2   = (const float*)d_in[30];
    const float* pb2   = (const float*)d_in[31];
    const float* hW1   = (const float*)d_in[32];
    const float* hb1   = (const float*)d_in[33];
    const float* hW2   = (const float*)d_in[34];
    const float* hb2   = (const float*)d_in[35];
    const float* hW3   = (const float*)d_in[36];
    const float* hb3   = (const float*)d_in[37];
    float* out = (float*)d_out;

    // workspace bump allocator (256B aligned)
    char* wsp = (char*)d_ws;
    size_t off = 0;
    auto alloc = [&](size_t bytes) -> void* {
        void* p = wsp + off;
        off += (bytes + 255) & ~(size_t)255;
        return p;
    };
    float*  h     = (float*)alloc((size_t)NN * HH * 4);
    float*  agg   = (float*)alloc((size_t)NN * HH * 4);   // also holds [N,32] layer-0 agg
    float*  t1    = (float*)alloc((size_t)NN * HH * 4);
    __bf16* W1_0t = (__bf16*)alloc((size_t)256 * 32 * 2);
    __bf16* W2_0t = (__bf16*)alloc((size_t)256 * 256 * 2);
    __bf16* W1_rt = (__bf16*)alloc((size_t)3 * 256 * 256 * 2);
    __bf16* W2_rt = (__bf16*)alloc((size_t)3 * 256 * 256 * 2);
    __bf16* fpWt  = (__bf16*)alloc((size_t)256 * 2048 * 2);
    __bf16* pW1t  = (__bf16*)alloc((size_t)256 * 1280 * 2);
    __bf16* pW2t  = (__bf16*)alloc((size_t)256 * 256 * 2);
    __bf16* hW1t  = (__bf16*)alloc((size_t)256 * 768 * 2);
    __bf16* hW2t  = (__bf16*)alloc((size_t)128 * 256 * 2);
    float*  sums  = (float*)alloc((size_t)BB * HH * 4);
    float*  cnts  = (float*)alloc((size_t)BB * 4);
    float*  z     = (float*)alloc((size_t)BB * 768 * 4);
    float*  tp    = (float*)alloc((size_t)BB * HH * 4);
    float*  z1    = (float*)alloc((size_t)BB * HH * 4);
    float*  z2    = (float*)alloc((size_t)BB * 128 * 4);

    auto cdiv = [](long a, long b) { return (int)((a + b - 1) / b); };
    auto conv = [&](const float* W, __bf16* Wt, int K, int Kpad, int Nout) {
        k_wt<<<cdiv((long)Nout * Kpad, 256), 256, 0, stream>>>(W, Wt, K, Kpad, Nout);
    };

    // --- one-time (per call) weight transpose/convert to bf16 ---
    conv(W1_0, W1_0t, 29, 32, 256);
    conv(W2_0, W2_0t, 256, 256, 256);
    for (int i = 0; i < 3; ++i) {
        conv(W1_r + (size_t)i * 65536, W1_rt + (size_t)i * 65536, 256, 256, 256);
        conv(W2_r + (size_t)i * 65536, W2_rt + (size_t)i * 65536, 256, 256, 256);
    }
    conv(fpW, fpWt, 2048, 2048, 256);
    conv(pW1, pW1t, 1280, 1280, 256);
    conv(pW2, pW2t, 256, 256, 256);
    conv(hW1, hW1t, 768, 768, 256);
    conv(hW2, hW2t, 256, 256, 128);

    // --- GINE layer 0 (in_dim = 29, padded to 32) ---
    k_pad29<<<cdiv((long)NN * 32, 256), 256, 0, stream>>>(x, agg);      // agg = pad(x)
    k_edge0<<<EE / 8, 256, 0, stream>>>(x, ei, ea, We0, beo0, agg);     // + messages
    GEMM(32, agg, 32, NN, 32, W1_0t, 256, b1_0, g1_0, bt1_0, 1, t1, 256);
    GEMM(256, t1, 256, NN, 256, W2_0t, 256, b2_0, gbn0, bbn0, 1, h, 256);

    // --- GINE layers 1..3 (in_dim = 256) ---
    for (int i = 0; i < 3; ++i) {
        k_copy4<<<cdiv((long)NN * HH / 4, 256), 256, 0, stream>>>(
            (const float4*)h, (float4*)agg, (long)NN * HH / 4);
        k_edge<<<EE / 8, 256, 0, stream>>>(h, ei, ea, We_r + (size_t)i * EFD * HH,
                                           beo_r + (size_t)i * HH, agg);
        GEMM(256, agg, 256, NN, 256, W1_rt + (size_t)i * 65536, 256,
             b1_r + (size_t)i * HH, g1_r + (size_t)i * HH, bt1_r + (size_t)i * HH,
             1, t1, 256);
        GEMM(256, t1, 256, NN, 256, W2_rt + (size_t)i * 65536, 256,
             b2_r + (size_t)i * HH, gbn_r + (size_t)i * HH, bbn_r + (size_t)i * HH,
             1, h, 256);
    }

    // --- global mean pool -> z[:, 0:256] ---
    k_fill<<<cdiv((long)BB * HH, 256), 256, 0, stream>>>(sums, 0.0f, BB * HH);
    k_fill<<<cdiv((long)BB, 256), 256, 0, stream>>>(cnts, 0.0f, BB);
    k_pool_sum<<<NN / 8, 256, 0, stream>>>(h, batch, sums, cnts);
    k_pool_div<<<cdiv((long)BB * HH, 256), 256, 0, stream>>>(sums, cnts, z);

    // --- fingerprint branch -> z[:, 256:512] ---
    GEMM(256, fp, 2048, BB, 2048, fpWt, 256, fpb, nullptr, nullptr, 1, z + 256, 768);

    // --- protein branch -> z[:, 512:768] ---
    GEMM(256, te, 1280, BB, 1280, pW1t, 256, pb1, nullptr, nullptr, 1, tp, 256);
    GEMM(256, tp, 256, BB, 256, pW2t, 256, pb2, nullptr, nullptr, 0, z + 512, 768);

    // --- fusion head ---
    GEMM(256, z, 768, BB, 768, hW1t, 256, hb1, nullptr, nullptr, 1, z1, 256);
    GEMM(256, z1, 256, BB, 256, hW2t, 128, hb2, nullptr, nullptr, 1, z2, 128);
    k_final<<<BB / 8, 256, 0, stream>>>(z2, hW3, hb3, out);

    (void)in_sizes; (void)n_in; (void)out_size; (void)ws_size;
}
